// EEGConnectomeGNN_16664473109174
// MI455X (gfx1250) — matrix operand order repaired and checked by reading, not compiled, not run on hardware
//
#include <hip/hip_runtime.h>
#include <hip/hip_bf16.h>

#define HID 128

typedef __attribute__((ext_vector_type(16))) __bf16 bf16x16;
typedef __attribute__((ext_vector_type(8)))  float  f32x8;
typedef __attribute__((ext_vector_type(8)))  float  v8f;

// ---------------- utility ----------------
__global__ void zero_f32(float* __restrict__ p, long long n) {
    long long i = (long long)blockIdx.x * blockDim.x + threadIdx.x;
    if (i < n) p[i] = 0.0f;
}

// Convert W [128k x 128n] f32 row-major -> Wt [n][k] bf16 (transposed)
__global__ void conv_wt(const float* __restrict__ W, __bf16* __restrict__ Wt) {
    int idx = blockIdx.x * blockDim.x + threadIdx.x;   // 16384 threads
    if (idx >= HID * HID) return;
    int k = idx >> 7, n = idx & 127;
    Wt[n * HID + k] = (__bf16)W[k * HID + n];
}

// ---------------- layer 0 (feature dim 1) ----------------
__global__ void scatter0(const float* __restrict__ x, const int* __restrict__ ei,
                         const float* __restrict__ ea, float* agg0, int nedges) {
    int e = blockIdx.x * blockDim.x + threadIdx.x;
    if (e >= nedges) return;
    float v = x[ei[e]] + ea[e];
    atomicAdd(&agg0[ei[nedges + e]], v > 0.0f ? v : 0.0f);
}

// t[n,c] = relu((x[n]+agg0[n]) * W1_0[c] + b1_0[c])
__global__ void layer0_mlp1(const float* __restrict__ x, const float* __restrict__ agg0,
                            const float* __restrict__ W1, const float* __restrict__ b1,
                            float* __restrict__ t, int nnodes) {
    int idx = blockIdx.x * blockDim.x + threadIdx.x;
    if (idx >= nnodes * HID) return;
    int n = idx >> 7, c = idx & 127;
    float u = x[n] + agg0[n];
    float v = u * W1[c] + b1[c];
    t[idx] = v > 0.0f ? v : 0.0f;
}

// ---------------- edge scatter (HID features) ----------------
// one wave per edge, 4 features per lane
__global__ __launch_bounds__(256)
void scatter_edges(const float* __restrict__ h, const int* __restrict__ ei,
                   const float* __restrict__ ea, float* agg, int nedges) {
    int e = blockIdx.x * 8 + (threadIdx.x >> 5);
    if (e >= nedges) return;
    int lane = threadIdx.x & 31;
    int src = ei[e];
    int dst = ei[nedges + e];
    float eav = ea[e];
    const float4 hv = *(const float4*)(h + (size_t)src * HID + lane * 4);
    float4 m;
    m.x = fmaxf(hv.x + eav, 0.0f);
    m.y = fmaxf(hv.y + eav, 0.0f);
    m.z = fmaxf(hv.z + eav, 0.0f);
    m.w = fmaxf(hv.w + eav, 0.0f);
    float* d = agg + (size_t)dst * HID + lane * 4;
    atomicAdd(d + 0, m.x);
    atomicAdd(d + 1, m.y);
    atomicAdd(d + 2, m.z);
    atomicAdd(d + 3, m.w);
}

// ---------------- WMMA GEMM: Out = relu((A [+ A2]) @ W + bias) ----------------
// A,A2,Out: [nrows,128] f32 row-major.  Wt: [n][k] bf16 (transposed weights).
// Block = 8 waves; wave w owns rows r0..r0+15 and all 8 column subtiles.
// HAS_A2 templated so the fused-add variant has no exec-mask branch in the
// k-loop and the plain variant drops the loads entirely.
template <bool HAS_A2>
__global__ __launch_bounds__(256)
void gemm128_bias_relu(const float* __restrict__ A, const float* A2,
                       const __bf16* __restrict__ Wt, const float* __restrict__ bias,
                       float* Out, int nrows) {
    const int wave = threadIdx.x >> 5;
    const int lane = threadIdx.x & 31;
    const int lh   = lane & 15;
    const int hi   = lane >> 4;
    const int r0   = blockIdx.x * 128 + wave * 16;

    v8f acc[8] = {};
    int arow = r0 + lh;
    if (arow >= nrows) arow = nrows - 1;   // clamp; OOB rows never stored
    const float* ap  = A + (size_t)arow * HID;
    const float* ap2 = HAS_A2 ? A2 + (size_t)arow * HID : nullptr;

    #pragma unroll
    for (int kc = 0; kc < HID; kc += 32) {
        // A fragment: lane half hi covers k = kb..kb+15 (contiguous)
        const int kb = kc + hi * 16;
        f32x8 g0 = *(const f32x8*)(ap + kb);
        f32x8 g1 = *(const f32x8*)(ap + kb + 8);
        if (HAS_A2) {
            g0 += *(const f32x8*)(ap2 + kb);
            g1 += *(const f32x8*)(ap2 + kb + 8);
        }
        bf16x16 a;
        #pragma unroll
        for (int i = 0; i < 8; ++i) {
            a[i]     = (__bf16)g0[i];
            a[i + 8] = (__bf16)g1[i];
        }
        // B fragment: column n = nt*16+lh, k = kc + hi*16 .. +15 (contiguous in Wt)
        const __bf16* wbase = Wt + (size_t)lh * HID + kc + hi * 16;
        #pragma unroll
        for (int nt = 0; nt < 8; ++nt) {
            bf16x16 b = *(const bf16x16*)(wbase + (size_t)nt * 16 * HID);
            acc[nt] = __builtin_amdgcn_wmma_f32_16x16x32_bf16(
                          false, a, false, b, (short)0, acc[nt], false, false);
        }
    }

    // D layout: lane holds column lh; VGPR v -> row r0 + hi*8 + v
    #pragma unroll
    for (int nt = 0; nt < 8; ++nt) {
        const int col = nt * 16 + lh;
        const float bv = bias[col];
        #pragma unroll
        for (int v = 0; v < 8; ++v) {
            const int row = r0 + hi * 8 + v;
            if (row < nrows) {
                float val = acc[nt][v] + bv;
                Out[(size_t)row * HID + col] = val > 0.0f ? val : 0.0f;
            }
        }
    }
}

// ---------------- pooling + classifier ----------------
__global__ __launch_bounds__(128)
void pool_sum(const float* __restrict__ h, const int* __restrict__ batch,
              float* pool, float* cnt, int nnodes) {
    int n = blockIdx.x;
    if (n >= nnodes) return;
    int c = threadIdx.x;
    int g = batch[n];
    atomicAdd(&pool[(size_t)g * HID + c], h[(size_t)n * HID + c]);
    if (c == 0) atomicAdd(&cnt[g], 1.0f);
}

__global__ __launch_bounds__(128)
void classify(const float* __restrict__ pool, const float* __restrict__ cnt,
              const float* __restrict__ Wc, const float* __restrict__ bc,
              float* __restrict__ out, int ncls) {
    int g = blockIdx.x;
    __shared__ float row[HID];
    float cg = cnt[g];
    float inv = 1.0f / (cg > 1.0f ? cg : 1.0f);
    row[threadIdx.x] = pool[(size_t)g * HID + threadIdx.x] * inv;
    __syncthreads();
    if (threadIdx.x < ncls) {
        float s = bc[threadIdx.x];
        for (int k = 0; k < HID; ++k) s += row[k] * Wc[k * ncls + threadIdx.x];
        out[g * ncls + threadIdx.x] = s;
    }
}

// ---------------- driver ----------------
extern "C" void kernel_launch(void* const* d_in, const int* in_sizes, int n_in,
                              void* d_out, int out_size, void* d_ws, size_t ws_size,
                              hipStream_t stream) {
    const float* x     = (const float*)d_in[0];
    const int*   ei    = (const int*)d_in[1];
    const float* ea    = (const float*)d_in[2];
    const int*   batch = (const int*)d_in[3];
    const float* W1_0  = (const float*)d_in[4];
    const float* b1_0  = (const float*)d_in[5];
    const float* W2_0  = (const float*)d_in[6];
    const float* b2_0  = (const float*)d_in[7];
    const float* W1_s  = (const float*)d_in[8];
    const float* b1_s  = (const float*)d_in[9];
    const float* W2_s  = (const float*)d_in[10];
    const float* b2_s  = (const float*)d_in[11];
    const float* Wc    = (const float*)d_in[12];
    const float* bc    = (const float*)d_in[13];
    float* out = (float*)d_out;

    const int nnodes  = in_sizes[0];          // x is [N,1]
    const int nedges  = in_sizes[2];          // edge_attr is [E,1]
    const int ncls    = in_sizes[13];         // bc
    const int ngraphs = out_size / ncls;

    // workspace layout
    float* h    = (float*)d_ws;                        // [N,128]
    float* agg  = h + (size_t)nnodes * HID;            // [N,128]
    float* agg0 = agg + (size_t)nnodes * HID;          // [N]
    float* pool = agg0 + nnodes;                       // [G,128]
    float* cnt  = pool + (size_t)ngraphs * HID;        // [G]
    __bf16* wt  = (__bf16*)(cnt + ngraphs);            // [128,128] bf16 (transposed)

    const long long nfeat = (long long)nnodes * HID;
    const dim3 z256(256);
    const dim3 gemm_grid((nnodes + 127) / 128);
    const dim3 convw_grid((HID * HID + 255) / 256);

    // ---- layer 0 (F=1) ----
    zero_f32<<<dim3((nnodes + 255) / 256), z256, 0, stream>>>(agg0, nnodes);
    scatter0<<<dim3((nedges + 255) / 256), z256, 0, stream>>>(x, ei, ea, agg0, nedges);
    layer0_mlp1<<<dim3((unsigned)((nfeat + 255) / 256)), z256, 0, stream>>>(
        x, agg0, W1_0, b1_0, agg, nnodes);
    conv_wt<<<convw_grid, z256, 0, stream>>>(W2_0, wt);
    gemm128_bias_relu<false><<<gemm_grid, z256, 0, stream>>>(
        agg, nullptr, wt, b2_0, h, nnodes);

    // ---- layers 1..3 (F=128) ----
    for (int i = 0; i < 3; ++i) {
        zero_f32<<<dim3((unsigned)((nfeat + 255) / 256)), z256, 0, stream>>>(agg, nfeat);
        scatter_edges<<<dim3((nedges + 7) / 8), z256, 0, stream>>>(h, ei, ea, agg, nedges);
        conv_wt<<<convw_grid, z256, 0, stream>>>(W1_s + (size_t)i * HID * HID, wt);
        gemm128_bias_relu<true><<<gemm_grid, z256, 0, stream>>>(
            h, agg, wt, b1_s + i * HID, agg, nnodes);          // in-place over agg
        conv_wt<<<convw_grid, z256, 0, stream>>>(W2_s + (size_t)i * HID * HID, wt);
        gemm128_bias_relu<false><<<gemm_grid, z256, 0, stream>>>(
            agg, nullptr, wt, b2_s + i * HID, h, nnodes);
    }

    // ---- global mean pool + classifier ----
    zero_f32<<<dim3((ngraphs * HID + ngraphs + 255) / 256), z256, 0, stream>>>(
        pool, (long long)ngraphs * HID + ngraphs);             // pool + cnt contiguous
    pool_sum<<<dim3(nnodes), dim3(HID), 0, stream>>>(h, batch, pool, cnt, nnodes);
    classify<<<dim3(ngraphs), dim3(HID), 0, stream>>>(pool, cnt, Wc, bc, out, ncls);
}
